// UncertaintyWeightedContrastiveLoss_29712583754308
// MI455X (gfx1250) — matrix operand order repaired and checked by reading, not compiled, not run on hardware
//
#include <hip/hip_runtime.h>
#include <math.h>

typedef __attribute__((ext_vector_type(16))) _Float16 v16h;
typedef __attribute__((ext_vector_type(8)))  _Float16 v8h;
typedef __attribute__((ext_vector_type(8)))  float    v8f;

#define NB        8192
#define NCLS      57
#define EDIM      256
#define TEMP_INV  (1.0f / 0.07f)
#define L_CONTRA  0.3f
#define L_CLS     0.7f

// LDS row stride in halves: 256 data + 8 pad -> 528B = 132 dwords,
// consecutive rows 4 banks apart -> conflict-free ds_load_b128 across 16 lanes.
#define BSTRIDE   264

// gfx1250 async global->LDS copy of 16 bytes (ASYNCcnt-tracked, no VGPR data).
__device__ __forceinline__ void async_cp16(const _Float16* g, _Float16* l) {
  // generic LDS pointer: addr[63:32]=shared aperture, addr[31:0]=LDS offset
  unsigned loff = (unsigned)(uintptr_t)l;
  asm volatile("global_load_async_to_lds_b128 %0, %1, off"
               :: "v"(loff), "v"(g) : "memory");
}
__device__ __forceinline__ void wait_async0() {
  asm volatile("s_wait_asynccnt 0x0" ::: "memory");
}

// ---------------------------------------------------------------- init
__global__ void init_ws(float* neg_sum, int* hist, float* cls_sum,
                        float* pair_sum, int* count) {
  int i = blockIdx.x * blockDim.x + threadIdx.x;
  if (i < NB) neg_sum[i] = 0.0f;
  if (i < NCLS) hist[i] = 0;
  if (i == 0) { *cls_sum = 0.0f; *pair_sum = 0.0f; *count = 0; }
}

// ---------------------------------------------------- classification + entropy
// one wave32 per row of 57 logits; lane covers class {lane, lane+32}
__global__ __launch_bounds__(256) void cls_entropy(
    const float* __restrict__ logits, const int* __restrict__ labels,
    float* __restrict__ u, float* cls_sum) {
  int lane = threadIdx.x & 31;
  int row  = blockIdx.x * 8 + (threadIdx.x >> 5);
  const float* lp = logits + (size_t)row * NCLS;
  float x1 = lp[lane];
  float x2 = (lane + 32 < NCLS) ? lp[lane + 32] : -INFINITY;
  float mx = fmaxf(x1, x2);
  #pragma unroll
  for (int m = 16; m >= 1; m >>= 1) mx = fmaxf(mx, __shfl_xor(mx, m, 32));
  float s = expf(x1 - mx) + ((lane + 32 < NCLS) ? expf(x2 - mx) : 0.0f);
  #pragma unroll
  for (int m = 16; m >= 1; m >>= 1) s += __shfl_xor(s, m, 32);
  float logZ = mx + logf(s);
  float p1 = expf(x1 - logZ);
  float ent = -p1 * logf(p1 + 1e-8f);
  if (lane + 32 < NCLS) {
    float p2 = expf(x2 - logZ);
    ent -= p2 * logf(p2 + 1e-8f);
  }
  #pragma unroll
  for (int m = 16; m >= 1; m >>= 1) ent += __shfl_xor(ent, m, 32);
  if (lane == 0) {
    u[row] = ent / logf((float)NCLS);
    int lbl = labels[row];
    atomicAdd(cls_sum, logZ - lp[lbl]);   // -log p[label]
  }
}

// ---------------------------------------------------------------- histogram
__global__ void task_hist(const int* __restrict__ task, int* hist) {
  int i = blockIdx.x * blockDim.x + threadIdx.x;
  if (i < NB) atomicAdd(&hist[task[i]], 1);
}

// ------------------------------------------- normalize + convert to f16
__global__ __launch_bounds__(256) void normalize_f16(
    const float* __restrict__ emb, _Float16* __restrict__ ef16) {
  int lane = threadIdx.x & 31;
  int row  = blockIdx.x * 8 + (threadIdx.x >> 5);
  const float* rp = emb + (size_t)row * EDIM + lane * 8;
  float4 a = *(const float4*)rp;
  float4 b = *(const float4*)(rp + 4);
  float ss = a.x*a.x + a.y*a.y + a.z*a.z + a.w*a.w
           + b.x*b.x + b.y*b.y + b.z*b.z + b.w*b.w;
  #pragma unroll
  for (int m = 16; m >= 1; m >>= 1) ss += __shfl_xor(ss, m, 32);
  float inv = 1.0f / fmaxf(sqrtf(ss), 1e-12f);
  v8h o;
  o[0] = (_Float16)(a.x*inv); o[1] = (_Float16)(a.y*inv);
  o[2] = (_Float16)(a.z*inv); o[3] = (_Float16)(a.w*inv);
  o[4] = (_Float16)(b.x*inv); o[5] = (_Float16)(b.y*inv);
  o[6] = (_Float16)(b.z*inv); o[7] = (_Float16)(b.w*inv);
  *(v8h*)(ef16 + (size_t)row * EDIM + lane * 8) = o;
}

// --------------------------------------------- validity weights + pair count
__global__ void valid_w(const int* __restrict__ task, const int* __restrict__ hist,
                        const float* __restrict__ u, float* __restrict__ w,
                        int* count) {
  int i = blockIdx.x * blockDim.x + threadIdx.x;
  if (i >= NB) return;
  int h = hist[task[i]];
  int pos = h - 1, neg = NB - h;
  bool valid = (pos > 0) && (neg > 0);
  w[i] = valid ? u[i] : 0.0f;
  if (valid) atomicAdd(count, pos);
}

// ---------------------------------------------------------------- WMMA GEMM
// Block = 128x128 output tile, 8 waves each owning 16(M) x 128(N).
// B panel (128 rows of e-f16) staged once per block into LDS via async copies;
// A rows are per-wave-unique and stream from global (single fixed row pointer).
template <int PASS>
__global__ __launch_bounds__(256) void sim_pass(
    const _Float16* __restrict__ ef16, const int* __restrict__ task,
    float* __restrict__ neg_sum, const float* __restrict__ w,
    float* pair_sum) {
  __shared__ _Float16 smem[128 * BSTRIDE];   // 66 KB

  const int lane = threadIdx.x & 31;
  const int wave = threadIdx.x >> 5;
  const int r    = lane & 15;
  const int hi   = lane >> 4;
  const int m0   = blockIdx.y * 128 + wave * 16;
  const int n0   = blockIdx.x * 128;

  // ---- stage B panel: 128 rows x 512B; wave copies rows {wave+8*it},
  // lane = 16B chunk -> fully coalesced 512B async transactions.
  {
    const _Float16* gB = ef16 + (size_t)n0 * EDIM;
    #pragma unroll
    for (int it = 0; it < 16; ++it) {
      int row = wave + 8 * it;
      async_cp16(gB + (size_t)row * EDIM + lane * 8,
                 &smem[row * BSTRIDE + lane * 8]);
    }
    wait_async0();
    __syncthreads();
  }

  const _Float16* rowA = ef16 + (size_t)(m0 + r) * EDIM;
  unsigned boff[8];
  #pragma unroll
  for (int t = 0; t < 8; ++t) boff[t] = (t * 16 + r) * BSTRIDE + 8 * hi;

  v8f acc[8];
  #pragma unroll
  for (int t = 0; t < 8; ++t) acc[t] = (v8f)0.0f;

  for (int k0 = 0; k0 < EDIM; k0 += 32) {
    // A 16x32 f16: VGPR0-3 = K[8*hi .. +7], VGPR4-7 = K[16+8*hi .. +7]
    v8h alo = *(const v8h*)(rowA + k0 + 8 * hi);
    v8h ahi = *(const v8h*)(rowA + k0 + 16 + 8 * hi);
    v16h a  = __builtin_shufflevector(alo, ahi,
                0,1,2,3,4,5,6,7,8,9,10,11,12,13,14,15);
    #pragma unroll
    for (int t = 0; t < 8; ++t) {
      // B 32x16 (KxN): lane holds column n=r, K = 8*hi .. +7 and 16+8*hi .. +7
      v8h b0 = *(const v8h*)&smem[boff[t] + k0];
      v8h b1 = *(const v8h*)&smem[boff[t] + k0 + 16];
      v16h b = __builtin_shufflevector(b0, b1,
                 0,1,2,3,4,5,6,7,8,9,10,11,12,13,14,15);
      acc[t] = __builtin_amdgcn_wmma_f32_16x16x32_f16(
                 false, a, false, b, (short)0, acc[t], false, false);
    }
  }

  // element (M = v + 8*hi, N = lane&15) in acc[t][v]; i = m0+M, j = n0+16t+N
  int ti[8];
  #pragma unroll
  for (int v = 0; v < 8; ++v) ti[v] = task[m0 + 8 * hi + v];

  if (PASS == 1) {
    float rsum[8];
    #pragma unroll
    for (int v = 0; v < 8; ++v) rsum[v] = 0.0f;
    #pragma unroll
    for (int t = 0; t < 8; ++t) {
      int j  = n0 + t * 16 + r;
      int tj = task[j];
      #pragma unroll
      for (int v = 0; v < 8; ++v) {
        // sim/T bounded by ~|14.3| -> hardware v_exp_f32 is safe (no
        // overflow/denorm fixup needed), and TRANS ops co-execute with WMMA.
        if (tj != ti[v])   // i==j implies same task -> auto-excluded
          rsum[v] += __expf(acc[t][v] * TEMP_INV);
      }
    }
    #pragma unroll
    for (int v = 0; v < 8; ++v) {
      float x = rsum[v];
      #pragma unroll
      for (int m = 8; m >= 1; m >>= 1) x += __shfl_xor(x, m, 32);
      if (r == 0) atomicAdd(&neg_sum[m0 + 8 * hi + v], x);
    }
  } else {
    float Ci[8], wi[8];
    #pragma unroll
    for (int v = 0; v < 8; ++v) {
      int i = m0 + 8 * hi + v;
      Ci[v] = neg_sum[i] + 1e-8f;
      wi[v] = w[i];
    }
    float psum = 0.0f;
    #pragma unroll
    for (int t = 0; t < 8; ++t) {
      int j  = n0 + t * 16 + r;
      int tj = task[j];
      #pragma unroll
      for (int v = 0; v < 8; ++v) {
        int i = m0 + 8 * hi + v;
        if (tj == ti[v] && i != j) {
          float s = acc[t][v] * TEMP_INV;
          psum += wi[v] * log1pf(Ci[v] * __expf(-s));
        }
      }
    }
    #pragma unroll
    for (int m = 16; m >= 1; m >>= 1) psum += __shfl_xor(psum, m, 32);
    if (lane == 0) atomicAdd(pair_sum, psum);
  }
}

// ---------------------------------------------------------------- finalize
__global__ void finalize(const float* cls_sum, const float* pair_sum,
                         const int* count, float* out) {
  float cls = *cls_sum / (float)NB;
  int cnt = *count;
  float contra = (cnt > 0) ? (*pair_sum / (float)cnt) : 0.0f;
  out[0] = L_CLS * cls + L_CONTRA * contra;
}

extern "C" void kernel_launch(void* const* d_in, const int* in_sizes, int n_in,
                              void* d_out, int out_size, void* d_ws, size_t ws_size,
                              hipStream_t stream) {
  const float* logits = (const float*)d_in[0];
  const float* emb    = (const float*)d_in[1];
  const int*   labels = (const int*)d_in[2];
  const int*   task   = (const int*)d_in[3];
  float* out = (float*)d_out;

  char* ws = (char*)d_ws;
  _Float16* ef16   = (_Float16*)ws;                                // 4 MB
  float* neg_sum   = (float*)(ws + (size_t)NB * EDIM * 2);         // 32 KB
  float* u         = neg_sum + NB;
  float* w         = u + NB;
  int*   hist      = (int*)(w + NB);
  float* cls_sum   = (float*)(hist + 64);
  float* pair_sum  = cls_sum + 1;
  int*   count     = (int*)(pair_sum + 1);

  init_ws<<<(NB + 255) / 256, 256, 0, stream>>>(neg_sum, hist, cls_sum, pair_sum, count);
  task_hist<<<(NB + 255) / 256, 256, 0, stream>>>(task, hist);
  cls_entropy<<<NB / 8, 256, 0, stream>>>(logits, labels, u, cls_sum);
  normalize_f16<<<NB / 8, 256, 0, stream>>>(emb, ef16);
  valid_w<<<(NB + 255) / 256, 256, 0, stream>>>(task, hist, u, w, count);

  dim3 grid(NB / 128, NB / 128);
  sim_pass<1><<<grid, 256, 0, stream>>>(ef16, task, neg_sum, w, pair_sum);
  sim_pass<2><<<grid, 256, 0, stream>>>(ef16, task, neg_sum, w, pair_sum);

  finalize<<<1, 1, 0, stream>>>(cls_sum, pair_sum, count, out);
}
